// DeltaNetLayer_22488448762124
// MI455X (gfx1250) — matrix operand order repaired and checked by reading, not compiled, not run on hardware
//
#include <hip/hip_runtime.h>
#include <hip/hip_bf16.h>

// DeltaNet-style gated linear attention layer for gfx1250 (MI455X).
// rmsnorm -> bf16 convert -> q/k/v GEMMs (WMMA bf16, double-buffered
// async-LDS staging, elu+1 epilogue) -> beta gate -> chunked WMMA scan
// (T=64, double-buffered async chunk loads, bf16 output) -> out GEMM
// (+bias+residual).

#define BQ 4
#define SQ 2048
#define DM 512
#define HQ 8
#define DH 64
#define ROWS (BQ * SQ)          // 8192
#define CHUNK 64
#define NCHUNK (SQ / CHUNK)     // 32
#define BETA_MIN 0.01f
#define BETA_MAX 0.995f
#define EPS_DELTA 1e-6f
#define EPS_RMS 1e-5f

typedef __attribute__((ext_vector_type(16))) __bf16 v16bf;
typedef __attribute__((ext_vector_type(8)))  float  v8f;

static __device__ __forceinline__ v8f vzero8() {
  v8f z;
#pragma unroll
  for (int i = 0; i < 8; ++i) z[i] = 0.0f;
  return z;
}

static __device__ __forceinline__ float clampf(float x, float lo, float hi) {
  return fminf(fmaxf(x, lo), hi);
}

// ---- gfx1250 async global->LDS copy (ASYNCcnt) ----
// LDS operand = low 32 bits of the flat LDS pointer (aperture maps addr[31:0]).
static __device__ __forceinline__ unsigned lds_addr32(const void* p) {
  return (unsigned)(unsigned long long)p;
}
static __device__ __forceinline__ void async_ld_b128(unsigned ldsa, const void* g) {
  asm volatile("global_load_async_to_lds_b128 %0, %1, off"
               :: "v"(ldsa), "v"((unsigned long long)g) : "memory");
}
// Wait until at most N async copies from this wave remain outstanding.
#define WAIT_ASYNC_LE(n) asm volatile("s_wait_asynccnt %0" :: "i"(n) : "memory")

// ---- WMMA fragment builders (16x16x32 bf16, wave32 layouts per CDNA5 ISA) ----
// A (16x32): lane L holds row M=L%16; VGPR j: K = {2j | 16+2(j-4)} + (L>=16 ? 8 : 0)
template <typename T>
static __device__ __forceinline__ v16bf
load_a_frag(const T* A, int mbase, int kbase, int ld, int lane) {
  const int M = mbase + (lane & 15);
  const int koff = (lane & 16) ? 8 : 0;
  v16bf f;
#pragma unroll
  for (int j = 0; j < 8; ++j) {
    const int kb = (j < 4) ? (2 * j) : (16 + 2 * (j - 4));
    const int k = kbase + kb + koff;
    f[2 * j]     = (__bf16)A[(size_t)M * ld + k];
    f[2 * j + 1] = (__bf16)A[(size_t)M * ld + k + 1];
  }
  return f;
}

// A with per-row scale: A'[M][K] = A[M][K] * rs[M]
template <typename T>
static __device__ __forceinline__ v16bf
load_a_frag_rowscale(const T* A, const float* rs, int mbase, int kbase, int ld, int lane) {
  const int M = mbase + (lane & 15);
  const int koff = (lane & 16) ? 8 : 0;
  const float s = rs[M];
  v16bf f;
#pragma unroll
  for (int j = 0; j < 8; ++j) {
    const int kb = (j < 4) ? (2 * j) : (16 + 2 * (j - 4));
    const int k = kbase + kb + koff;
    f[2 * j]     = (__bf16)((float)A[(size_t)M * ld + k] * s);
    f[2 * j + 1] = (__bf16)((float)A[(size_t)M * ld + k + 1] * s);
  }
  return f;
}

// A from transposed source with per-k scale: A'[M][K] = S[K][M] * ks[K]
template <typename T>
static __device__ __forceinline__ v16bf
load_aT_frag_kscale(const T* Smem, const float* ks, int mbase, int kbase, int ld, int lane) {
  const int M = mbase + (lane & 15);
  const int koff = (lane & 16) ? 8 : 0;
  v16bf f;
#pragma unroll
  for (int j = 0; j < 8; ++j) {
    const int kb = (j < 4) ? (2 * j) : (16 + 2 * (j - 4));
    const int k = kbase + kb + koff;
    f[2 * j]     = (__bf16)((float)Smem[(size_t)k * ld + M] * ks[k]);
    f[2 * j + 1] = (__bf16)((float)Smem[(size_t)(k + 1) * ld + M] * ks[k + 1]);
  }
  return f;
}

// B (32x16) row-major source: B[k][n] = S[k*ld + nbase + n]
// lane L holds col N=L%16; VGPR j: K = {2j | 16+2(j-4)} + (L>=16 ? 8 : 0)
template <typename T>
static __device__ __forceinline__ v16bf
load_b_frag(const T* Smem, int nbase, int kbase, int ld, int lane) {
  const int N = nbase + (lane & 15);
  const int koff = (lane & 16) ? 8 : 0;
  v16bf f;
#pragma unroll
  for (int j = 0; j < 8; ++j) {
    const int k = kbase + ((j < 4) ? (2 * j) : (16 + 2 * (j - 4))) + koff;
    f[2 * j]     = (__bf16)Smem[(size_t)k * ld + N];
    f[2 * j + 1] = (__bf16)Smem[(size_t)(k + 1) * ld + N];
  }
  return f;
}

// B from transposed source: B[k][n] = S[(nbase+n)*ld + kbase + k]
template <typename T>
static __device__ __forceinline__ v16bf
load_bT_frag(const T* Smem, int nbase, int kbase, int ld, int lane) {
  const int N = nbase + (lane & 15);
  const int koff = (lane & 16) ? 8 : 0;
  v16bf f;
#pragma unroll
  for (int j = 0; j < 8; ++j) {
    const int k = kbase + ((j < 4) ? (2 * j) : (16 + 2 * (j - 4))) + koff;
    f[2 * j]     = (__bf16)Smem[(size_t)N * ld + k];
    f[2 * j + 1] = (__bf16)Smem[(size_t)N * ld + k + 1];
  }
  return f;
}

// C/D (16x16 f32): lane L: N=L%16; VGPR i: M = i + (L>=16 ? 8 : 0)
static __device__ __forceinline__ void
store_c(float* C, int mbase, int nbase, int ld, v8f acc, int lane) {
  const int n = nbase + (lane & 15);
  const int moff = (lane & 16) ? 8 : 0;
#pragma unroll
  for (int i = 0; i < 8; ++i)
    C[(size_t)(mbase + moff + i) * ld + n] = acc[i];
}

static __device__ __forceinline__ v8f wmma_bf16(v16bf a, v16bf b, v8f c) {
  return __builtin_amdgcn_wmma_f32_16x16x32_bf16(false, a, false, b, (short)0, c,
                                                 false, false);
}

// ------------------------------ RMSNorm ------------------------------
__global__ __launch_bounds__(256) void rmsnorm_kernel(const float* __restrict__ x,
                                                      const float* __restrict__ w,
                                                      float* __restrict__ h) {
  __shared__ float red[256];
  const int row = blockIdx.x;
  const int tid = threadIdx.x;
  const size_t base = (size_t)row * DM;
  float ss = 0.f;
  for (int i = tid; i < DM; i += 256) {
    float v = x[base + i];
    ss += v * v;
  }
  red[tid] = ss;
  __syncthreads();
  for (int s = 128; s > 0; s >>= 1) {
    if (tid < s) red[tid] += red[tid + s];
    __syncthreads();
  }
  const float inv = rsqrtf(red[0] / (float)DM + EPS_RMS);
  for (int i = tid; i < DM; i += 256) h[base + i] = x[base + i] * inv * w[i];
}

// ------------------------------ f32 -> bf16 convert ------------------------------
__global__ __launch_bounds__(256) void cvt_bf16_kernel(const float* __restrict__ s,
                                                       __bf16* __restrict__ d, int n) {
  const int stride = gridDim.x * blockDim.x;
  for (int i = blockIdx.x * blockDim.x + threadIdx.x; i < n; i += stride)
    d[i] = (__bf16)s[i];
}

// --------------- GEMM: C = act(A @ W^T [+bias+resid]), bf16 inputs ---------------
// A: [M,K] bf16, W: [N,K] bf16. Double-buffered async-LDS staging:
// tile i+1's global_load_async_to_lds overlaps tile i's WMMA.
// ACT: 0=none, 1=elu(x)+1, 2=+bias[n]+resid[m*N+n]
template <int ACT>
__global__ __launch_bounds__(256) void gemm_bf16_kernel(const __bf16* __restrict__ A,
                                                        const __bf16* __restrict__ W,
                                                        float* __restrict__ C,
                                                        const float* __restrict__ bias,
                                                        const float* __restrict__ resid,
                                                        int M, int N, int K) {
  __shared__ __bf16 As[2][64 * 32];
  __shared__ __bf16 Ws[2][64 * 32];
  const int tid = threadIdx.x;
  const int lane = tid & 31;
  const int wave = tid >> 5;
  const int m0 = blockIdx.x * 64;
  const int n0 = blockIdx.y * 64;

  // wave's two tiles share the A row
  const int tm = (wave >> 1) * 16;
  const int tn0 = (wave & 1) * 32;
  const int tn1 = tn0 + 16;

  const int r = tid >> 2, c8 = (tid & 3) * 8;  // 16B per lane per matrix

  v8f acc0 = vzero8(), acc1 = vzero8();

  // prologue: stage tile 0 into buffer 0
  async_ld_b128(lds_addr32(&As[0][r * 32 + c8]), &A[(size_t)(m0 + r) * K + c8]);
  async_ld_b128(lds_addr32(&Ws[0][r * 32 + c8]), &W[(size_t)(n0 + r) * K + c8]);

  const int nk = K / 32;
  for (int i = 0; i < nk; ++i) {
    const int cur = i & 1;
    if (i + 1 < nk) {  // stage tile i+1 into the other buffer, then wait tile i
      const int kk = (i + 1) * 32;
      async_ld_b128(lds_addr32(&As[cur ^ 1][r * 32 + c8]),
                    &A[(size_t)(m0 + r) * K + kk + c8]);
      async_ld_b128(lds_addr32(&Ws[cur ^ 1][r * 32 + c8]),
                    &W[(size_t)(n0 + r) * K + kk + c8]);
      WAIT_ASYNC_LE(2);
    } else {
      WAIT_ASYNC_LE(0);
    }
    __syncthreads();
    v16bf af = load_a_frag(As[cur], tm, 0, 32, lane);
    v16bf b0 = load_bT_frag(Ws[cur], tn0, 0, 32, lane);  // B[k][n] = W[n][k]
    v16bf b1 = load_bT_frag(Ws[cur], tn1, 0, 32, lane);
    acc0 = wmma_bf16(af, b0, acc0);
    acc1 = wmma_bf16(af, b1, acc1);
    __syncthreads();  // protects buffer reuse by next iteration's copies
  }

#pragma unroll
  for (int t = 0; t < 2; ++t) {
    const v8f acc = t ? acc1 : acc0;
    const int gn = n0 + (t ? tn1 : tn0) + (lane & 15);
    const int moff = (lane & 16) ? 8 : 0;
#pragma unroll
    for (int i = 0; i < 8; ++i) {
      const int gm = m0 + tm + moff + i;
      float v = acc[i];
      if (ACT == 1) v = (v > 0.f) ? (v + 1.f) : __expf(v);  // elu(x)+1
      if (ACT == 2) v += bias[gn] + resid[(size_t)gm * N + gn];
      C[(size_t)gm * N + gn] = v;
    }
  }
}

// ------------------------------ Beta gate ------------------------------
__global__ __launch_bounds__(256) void beta_kernel(const float* __restrict__ h,
                                                   const float* __restrict__ Wb,
                                                   const float* __restrict__ bb,
                                                   const float* __restrict__ rec,
                                                   float* __restrict__ beta) {
  const int row = blockIdx.x;
  const int head = threadIdx.x >> 5;  // 8 waves -> 8 heads
  const int lane = threadIdx.x & 31;
  float acc = 0.f;
  for (int k = lane; k < DM; k += 32)
    acc += h[(size_t)row * DM + k] * Wb[(size_t)head * DM + k];
#pragma unroll
  for (int off = 16; off > 0; off >>= 1) acc += __shfl_xor(acc, off, 32);
  if (lane == 0) {
    float z = acc + bb[head] + rec[head];
    beta[(size_t)row * HQ + head] = 1.f / (1.f + __expf(-z));
  }
}

// ------------------------------ Chunked scan (T=64, WMMA) --------------------
// One workgroup (256 thr / 8 waves) per batch-head; state H1,H2,Z1,Z2 in LDS.
// q/k/v chunk buffers are double-buffered: chunk c+1's async HBM->LDS copies
// overlap all of chunk c's WMMA phases.
#define SC_Q0 0
#define SC_K0 4096
#define SC_V0 8192
#define SC_Q1 12288
#define SC_K1 16384
#define SC_V1 20480
#define SC_S  24576      // S_raw -> decayed S -> num (reused)
#define SC_H1 28672
#define SC_H2 32768
#define SC_Z1 36864
#define SC_Z2 36928
#define SC_L1 36992
#define SC_L2 37056
#define SC_EL1 37120
#define SC_EL2 37184
#define SC_W1 37248
#define SC_W2 37312
#define SC_MV 37376
#define SC_DEN 37440
#define SC_PART 37504
#define SC_LB1 37760
#define SC_LB2 37824
#define SC_FLOATS 37888  // 151552 bytes dynamic LDS (CDNA5: up to 320KB/WGP)

__global__ __launch_bounds__(256) void scan_kernel(const float* __restrict__ qb,
                                                   const float* __restrict__ kb,
                                                   const float* __restrict__ vb,
                                                   const float* __restrict__ beta,
                                                   const float* __restrict__ mask,
                                                   const float* __restrict__ pbb1,
                                                   const float* __restrict__ pbb2,
                                                   __bf16* __restrict__ ob) {
  extern __shared__ float lds[];
  float* Smat = lds + SC_S;
  float* H1 = lds + SC_H1;
  float* H2 = lds + SC_H2;
  float* Z1 = lds + SC_Z1;
  float* Z2 = lds + SC_Z2;
  float* L1 = lds + SC_L1;
  float* L2 = lds + SC_L2;
  float* eL1 = lds + SC_EL1;
  float* eL2 = lds + SC_EL2;
  float* w1 = lds + SC_W1;
  float* w2 = lds + SC_W2;
  float* mvec = lds + SC_MV;
  float* den = lds + SC_DEN;
  float* parts = lds + SC_PART;
  float* lb1 = lds + SC_LB1;
  float* lb2 = lds + SC_LB2;

  const int bh = blockIdx.x;
  const int b = bh / HQ, hd = bh % HQ;
  const int tid = threadIdx.x;
  const int lane = tid & 31;
  const int wave = tid >> 5;

  const int tm = (wave >> 1) * 16;   // wave tile row
  const int tn0 = (wave & 1) * 32;   // wave tile cols
  const int tn1 = tn0 + 16;

  const float b1_base = clampf(1.f / (1.f + __expf(-pbb1[0])), BETA_MIN, BETA_MAX);
  const float b2_base = clampf(1.f / (1.f + __expf(-pbb2[0])), BETA_MIN, BETA_MAX);

  for (int i = tid; i < DH * DH; i += 256) { H1[i] = 0.f; H2[i] = 0.f; }
  if (tid < DH) { Z1[tid] = 0.f; Z2[tid] = 0.f; }

  const size_t rowbase = ((size_t)b * SQ) * DM + (size_t)hd * DH;

  // issue 12 async b128 copies (q,k,v chunk) into buffer `buf`
  auto issue_chunk = [&](int buf, int cidx) {
    float* q_c = lds + (buf ? SC_Q1 : SC_Q0);
    float* k_c = lds + (buf ? SC_K1 : SC_K0);
    float* v_c = lds + (buf ? SC_V1 : SC_V0);
    const int s0 = cidx * CHUNK;
#pragma unroll
    for (int j = 0; j < 4; ++j) {
      const int g4 = tid + j * 256;              // float4 index 0..1023
      const int t = g4 >> 4, c4 = (g4 & 15) * 4;
      const size_t goff = rowbase + (size_t)(s0 + t) * DM + c4;
      async_ld_b128(lds_addr32(&q_c[g4 * 4]), qb + goff);
      async_ld_b128(lds_addr32(&k_c[g4 * 4]), kb + goff);
      async_ld_b128(lds_addr32(&v_c[g4 * 4]), vb + goff);
    }
  };

  issue_chunk(0, 0);  // prologue
  __syncthreads();    // H/Z init visible

  for (int c = 0; c < NCHUNK; ++c) {
    const int cur = c & 1;
    float* q_c = lds + (cur ? SC_Q1 : SC_Q0);
    float* k_c = lds + (cur ? SC_K1 : SC_K0);
    float* v_c = lds + (cur ? SC_V1 : SC_V0);
    const int s0 = c * CHUNK;

    // stage chunk c+1 into the other buffer; wait only for chunk c's copies
    if (c + 1 < NCHUNK) {
      issue_chunk(cur ^ 1, c + 1);
      WAIT_ASYNC_LE(12);
    } else {
      WAIT_ASYNC_LE(0);
    }
    // prefetch chunk c+2 into L2
    if (c + 2 < NCHUNK) {
      const size_t nb = rowbase + (size_t)(s0 + 2 * CHUNK + (tid & 63)) * DM;
      if (tid < 64) __builtin_prefetch(qb + nb, 0, 1);
      else if (tid < 128) __builtin_prefetch(kb + nb, 0, 1);
      else if (tid < 192) __builtin_prefetch(vb + nb, 0, 1);
    }
    if (tid < CHUNK) {
      const int t = tid;
      const float m = mask[(size_t)b * SQ + s0 + t];
      const float bt = beta[((size_t)b * SQ + s0 + t) * HQ + hd];
      const float bin = clampf(bt, BETA_MIN, BETA_MAX);
      lb1[t] = __logf(clampf(b1_base * bin, BETA_MIN, BETA_MAX));
      lb2[t] = __logf(clampf(b2_base * bin, BETA_MIN, BETA_MAX));
      mvec[t] = m;
    }
    __syncthreads();
    if (tid == 0) {
      float a1 = 0.f, a2 = 0.f;
      for (int t = 0; t < CHUNK; ++t) {
        a1 += lb1[t]; L1[t] = a1;
        a2 += lb2[t]; L2[t] = a2;
      }
    }
    __syncthreads();
    if (tid < CHUNK) {
      eL1[tid] = __expf(L1[tid]);
      eL2[tid] = __expf(L2[tid]);
      w1[tid] = __expf(L1[CHUNK - 1] - L1[tid]) * mvec[tid];
      w2[tid] = __expf(L2[CHUNK - 1] - L2[tid]) * mvec[tid];
    }
    __syncthreads();

    // --- Phase 1: S_raw = phi(Q) @ phi(K)^T ---
    {
      v8f a0 = vzero8(), a1 = vzero8();
#pragma unroll
      for (int kk = 0; kk < DH; kk += 32) {
        v16bf af = load_a_frag(q_c, tm, kk, DH, lane);
        v16bf b0 = load_bT_frag(k_c, tn0, kk, DH, lane);  // B[d][j] = k_c[j][d]
        v16bf b1 = load_bT_frag(k_c, tn1, kk, DH, lane);
        a0 = wmma_bf16(af, b0, a0);
        a1 = wmma_bf16(af, b1, a1);
      }
      store_c(Smat, tm, tn0, DH, a0, lane);
      store_c(Smat, tm, tn1, DH, a1, lane);
    }
    __syncthreads();

    // --- Phase 2: causal decay mask (both states fused) ---
    for (int i = tid; i < CHUNK * CHUNK; i += 256) {
      const int t = i >> 6, j = i & 63;
      float v = 0.f;
      if (j <= t)
        v = Smat[i] * mvec[j] *
            (__expf(L1[t] - L1[j]) + __expf(L2[t] - L2[j]));
      Smat[i] = v;
    }
    __syncthreads();

    // --- Phase 3: den[t] = rowsum(S) + eL1*q.Z1 + eL2*q.Z2 ---
    {
      const int t = tid >> 2, qd = tid & 3;
      float p = 0.f;
      for (int j = qd * 16; j < qd * 16 + 16; ++j) p += Smat[t * DH + j];
      parts[tid] = p;
    }
    __syncthreads();
    if (tid < CHUNK) {
      const int t = tid;
      float ds = parts[t * 4] + parts[t * 4 + 1] + parts[t * 4 + 2] + parts[t * 4 + 3];
      float d1 = 0.f, d2 = 0.f;
      for (int dk = 0; dk < DH; ++dk) {
        const float qv = q_c[t * DH + dk];
        d1 += qv * Z1[dk];
        d2 += qv * Z2[dk];
      }
      den[t] = ds + eL1[t] * d1 + eL2[t] * d2;
    }
    __syncthreads();

    // --- Phase 4: num = S @ V + (eL1.Q) @ H1 + (eL2.Q) @ H2 ---
    v8f n0acc = vzero8(), n1acc = vzero8();
#pragma unroll
    for (int kk = 0; kk < CHUNK; kk += 32) {
      v16bf af = load_a_frag(Smat, tm, kk, CHUNK, lane);
      v16bf b0 = load_b_frag(v_c, tn0, kk, DH, lane);
      v16bf b1 = load_b_frag(v_c, tn1, kk, DH, lane);
      n0acc = wmma_bf16(af, b0, n0acc);
      n1acc = wmma_bf16(af, b1, n1acc);
    }
#pragma unroll
    for (int kk = 0; kk < DH; kk += 32) {
      v16bf af = load_a_frag_rowscale(q_c, eL1, tm, kk, DH, lane);
      v16bf b0 = load_b_frag(H1, tn0, kk, DH, lane);
      v16bf b1 = load_b_frag(H1, tn1, kk, DH, lane);
      n0acc = wmma_bf16(af, b0, n0acc);
      n1acc = wmma_bf16(af, b1, n1acc);
    }
#pragma unroll
    for (int kk = 0; kk < DH; kk += 32) {
      v16bf af = load_a_frag_rowscale(q_c, eL2, tm, kk, DH, lane);
      v16bf b0 = load_b_frag(H2, tn0, kk, DH, lane);
      v16bf b1 = load_b_frag(H2, tn1, kk, DH, lane);
      n0acc = wmma_bf16(af, b0, n0acc);
      n1acc = wmma_bf16(af, b1, n1acc);
    }
    __syncthreads();  // all reads of Smat/H done -> safe to overwrite Smat
    store_c(Smat, tm, tn0, CHUNK, n0acc, lane);
    store_c(Smat, tm, tn1, CHUNK, n1acc, lane);
    __syncthreads();

    // --- Phase 5: output (bf16 for the final GEMM) + Z state update ---
    for (int i = tid; i < CHUNK * DH; i += 256) {
      const int t = i >> 6, dv = i & 63;
      const float o = Smat[i] / fmaxf(den[t], EPS_DELTA) * mvec[t];
      ob[rowbase + (size_t)(s0 + t) * DM + dv] = (__bf16)o;
    }
    if (tid < 128) {
      const int dk = tid & 63;
      float* Z = (tid < 64) ? Z1 : Z2;
      const float* w = (tid < 64) ? w1 : w2;
      const float eT = (tid < 64) ? __expf(L1[CHUNK - 1]) : __expf(L2[CHUNK - 1]);
      float a = 0.f;
      for (int j = 0; j < CHUNK; ++j) a += w[j] * k_c[j * DH + dk];
      Z[dk] = eT * Z[dk] + a;
    }
    __syncthreads();

    // --- Phase 6: H state update: H <- e^{L_T} H + (diag(w) K)^T @ V ---
    {
      const float e1T = __expf(L1[CHUNK - 1]);
      const float e2T = __expf(L2[CHUNK - 1]);
      v8f h10 = vzero8(), h11 = vzero8(), h20 = vzero8(), h21 = vzero8();
#pragma unroll
      for (int kk = 0; kk < CHUNK; kk += 32) {
        v16bf ak1 = load_aT_frag_kscale(k_c, w1, tm, kk, DH, lane);
        v16bf ak2 = load_aT_frag_kscale(k_c, w2, tm, kk, DH, lane);
        v16bf b0 = load_b_frag(v_c, tn0, kk, DH, lane);
        v16bf b1 = load_b_frag(v_c, tn1, kk, DH, lane);
        h10 = wmma_bf16(ak1, b0, h10);
        h11 = wmma_bf16(ak1, b1, h11);
        h20 = wmma_bf16(ak2, b0, h20);
        h21 = wmma_bf16(ak2, b1, h21);
      }
      const int moff = (lane & 16) ? 8 : 0;
#pragma unroll
      for (int i = 0; i < 8; ++i) {
        const int m = tm + moff + i;
        const int i0 = m * DH + tn0 + (lane & 15);
        const int i1 = m * DH + tn1 + (lane & 15);
        H1[i0] = e1T * H1[i0] + h10[i];
        H1[i1] = e1T * H1[i1] + h11[i];
        H2[i0] = e2T * H2[i0] + h20[i];
        H2[i1] = e2T * H2[i1] + h21[i];
      }
    }
    __syncthreads();  // protects q/k/v buffer reuse next chunk
  }
}

// ------------------------------ Launch ------------------------------
extern "C" void kernel_launch(void* const* d_in, const int* in_sizes, int n_in,
                              void* d_out, int out_size, void* d_ws, size_t ws_size,
                              hipStream_t stream) {
  (void)in_sizes; (void)n_in; (void)out_size; (void)ws_size;
  const float* x    = (const float*)d_in[0];
  const float* mask = (const float*)d_in[1];
  const float* Wq   = (const float*)d_in[2];
  const float* Wk   = (const float*)d_in[3];
  const float* Wv   = (const float*)d_in[4];
  const float* Wb   = (const float*)d_in[5];
  const float* bb   = (const float*)d_in[6];
  const float* Wo   = (const float*)d_in[7];
  const float* bo   = (const float*)d_in[8];
  const float* ln_w = (const float*)d_in[9];
  const float* rec  = (const float*)d_in[10];
  const float* bb1  = (const float*)d_in[11];
  const float* bb2  = (const float*)d_in[12];
  float* out = (float*)d_out;

  // Workspace carve (~86 MB), 16B-aligned blocks.
  const size_t per = (size_t)ROWS * DM;       // 4 Mi elements
  const size_t wsz = (size_t)DM * DM;         // 256 Ki elements
  float* h   = (float*)d_ws;
  float* q   = h + per;
  float* k   = q + per;
  float* v   = k + per;
  float* be  = v + per;
  float* fend = be + (size_t)ROWS * HQ;
  __bf16* h_bf  = (__bf16*)fend;
  __bf16* o_bf  = h_bf + per;
  __bf16* wq_bf = o_bf + per;
  __bf16* wk_bf = wq_bf + wsz;
  __bf16* wv_bf = wk_bf + wsz;
  __bf16* wo_bf = wv_bf + wsz;

  rmsnorm_kernel<<<ROWS, 256, 0, stream>>>(x, ln_w, h);

  cvt_bf16_kernel<<<1024, 256, 0, stream>>>(h, h_bf, (int)per);
  cvt_bf16_kernel<<<128, 256, 0, stream>>>(Wq, wq_bf, (int)wsz);
  cvt_bf16_kernel<<<128, 256, 0, stream>>>(Wk, wk_bf, (int)wsz);
  cvt_bf16_kernel<<<128, 256, 0, stream>>>(Wv, wv_bf, (int)wsz);
  cvt_bf16_kernel<<<128, 256, 0, stream>>>(Wo, wo_bf, (int)wsz);

  dim3 ggrid(ROWS / 64, DM / 64);
  gemm_bf16_kernel<1><<<ggrid, 256, 0, stream>>>(h_bf, wq_bf, q, nullptr, nullptr,
                                                 ROWS, DM, DM);  // phi(q)
  gemm_bf16_kernel<1><<<ggrid, 256, 0, stream>>>(h_bf, wk_bf, k, nullptr, nullptr,
                                                 ROWS, DM, DM);  // phi(k)
  gemm_bf16_kernel<0><<<ggrid, 256, 0, stream>>>(h_bf, wv_bf, v, nullptr, nullptr,
                                                 ROWS, DM, DM);

  beta_kernel<<<ROWS, 256, 0, stream>>>(h, Wb, bb, rec, be);

  scan_kernel<<<BQ * HQ, 256, SC_FLOATS * sizeof(float), stream>>>(
      q, k, v, be, mask, bb1, bb2, o_bf);

  gemm_bf16_kernel<2><<<ggrid, 256, 0, stream>>>(o_bf, wo_bf, out, bo, x,
                                                 ROWS, DM, DM);  // +bias+residual
}
